// Spectrogrammer_13855564497498
// MI455X (gfx1250) — matrix operand order repaired and checked by reading, not compiled, not run on hardware
//
#include <hip/hip_runtime.h>
#include <hip/hip_bf16.h>

typedef __attribute__((ext_vector_type(16))) _Float16 v16h;
typedef __attribute__((ext_vector_type(8)))  float    v8f;

#define D_TAPS   4096
#define KFILT    240
#define STRIDE_  32
#define NBATCH   16
#define TSIG     441000
#define LOUT     13654          // (TSIG - D_TAPS)/STRIDE_ + 1
#define NTILE_L  256            // frames per block = 8 waves * 32
#define NBLK_L   ((LOUT + NTILE_L - 1) / NTILE_L)   // 54
#define TPAD     446464         // >= 32*(NBLK_L*NTILE_L-1)+4096, mult of 256
#define MT       15             // 240 / 16 filter tiles
#define NCHUNK   128            // 4096 / 32 K-chunks
#define STAGE    4              // chunks per LDS stage
#define NSTAGES  (NCHUNK / STAGE)

#if __has_builtin(__builtin_amdgcn_global_load_async_to_lds_b128) && \
    __has_builtin(__builtin_amdgcn_s_wait_asynccnt)
#define USE_ASYNC 1
typedef int v4i_t __attribute__((vector_size(16)));
typedef __attribute__((address_space(1))) v4i_t gv4i_t;   // global (prints as __device__)
typedef __attribute__((address_space(3))) v4i_t lv4i_t;   // LDS    (prints as __shared__)
#else
#define USE_ASYNC 0
#endif

// ---------------------------------------------------------------------------
// Prep 1: fp32 signal -> fp16, zero-padded to TPAD per batch.
// ---------------------------------------------------------------------------
__global__ void prep_signal(const float* __restrict__ sig, _Float16* __restrict__ sigH) {
    long idx = (long)blockIdx.x * blockDim.x + threadIdx.x;
    const long total = (long)NBATCH * TPAD;
    if (idx >= total) return;
    int b = (int)(idx / TPAD);
    int t = (int)(idx - (long)b * TPAD);
    sigH[idx] = (t < TSIG) ? (_Float16)sig[(long)b * TSIG + t] : (_Float16)0.0f;
}

// ---------------------------------------------------------------------------
// Prep 2: fp32 filters -> fp16, pre-swizzled into WMMA A-matrix per-lane
// layout. filtH[set][m][c][lane*16 + j] with
//   M = lane&15, Kbase = (lane<16 ? 0 : 16)
//   K = Kbase + j
// ---------------------------------------------------------------------------
__global__ void prep_filters(const float* __restrict__ wsin,
                             const float* __restrict__ wcos,
                             _Float16* __restrict__ filtH) {
    int idx = blockIdx.x * blockDim.x + threadIdx.x;
    const int perSet = MT * NCHUNK * 512;
    if (idx >= 2 * perSet) return;
    int set  = idx / perSet;
    int r    = idx - set * perSet;
    int m    = r / (NCHUNK * 512);
    int r2   = r - m * (NCHUNK * 512);
    int c    = r2 >> 9;
    int h    = r2 & 511;
    int lane = h >> 4;
    int j    = h & 15;
    int M    = lane & 15;
    int kb   = (lane < 16) ? 0 : 16;
    int K    = kb + j;
    int k    = m * 16 + M;
    int d    = c * 32 + K;
    const float* w = set ? wcos : wsin;
    filtH[idx] = (_Float16)w[k * D_TAPS + d];
}

// ---------------------------------------------------------------------------
// Main: im2col GEMM on WMMA. Each wave: 16 filters x 32 frames, four f32
// accumulators (sin/cos x 2 N-tiles) -> every A fetch feeds 2 WMMAs.
// Filters double-buffered through LDS via async global->LDS (ASYNCcnt) when
// available; B tiles are 32 contiguous bytes of fp16 signal per lane.
// ---------------------------------------------------------------------------
__global__ __launch_bounds__(256) void spect_wmma(
    const _Float16* __restrict__ sigH,
    const _Float16* __restrict__ filtH,
    float* __restrict__ out) {

    __shared__ __align__(16) _Float16 lds[2][2][STAGE][512];   // 16 KB

    const int tid  = threadIdx.x;
    const int wave = tid >> 5;
    const int lane = tid & 31;
    const int m    = blockIdx.y;
    const int b    = blockIdx.z;
    const int l0   = blockIdx.x * NTILE_L + wave * 32;

    // B-matrix bases: column N = l0 + (lane&15) (+16 for tile 1),
    // K-run base = (lane>=16 ? 16 : 0). 16 contiguous halves per chunk.
    const _Float16* bbase0 =
        sigH + (size_t)b * TPAD + (size_t)STRIDE_ * (l0 + (lane & 15)) + ((lane >> 4) << 4);
    const _Float16* bbase1 = bbase0 + (size_t)STRIDE_ * 16;

    // LDS fill assignment: each thread moves 32 bytes per stage.
    const int sset  = tid >> 7;            // 0: sin, 1: cos
    const int scl   = (tid & 127) >> 5;    // chunk-in-stage
    const int slane = tid & 31;
    const _Float16* fbase =
        filtH + ((size_t)(sset * MT + m) * NCHUNK) * 512 + (size_t)slane * 16;

    v8f aS0 = {0.f,0.f,0.f,0.f,0.f,0.f,0.f,0.f};
    v8f aC0 = aS0, aS1 = aS0, aC1 = aS0;

#if USE_ASYNC
    auto issue_stage = [&](int s, int bufi) {
        const _Float16* gp = fbase + (size_t)(s * STAGE + scl) * 512;
        _Float16* lp = &lds[bufi][sset][scl][slane * 16];
        __builtin_amdgcn_global_load_async_to_lds_b128(
            (gv4i_t*)gp, (lv4i_t*)lp, 0, 0);
        __builtin_amdgcn_global_load_async_to_lds_b128(
            (gv4i_t*)(gp + 8), (lv4i_t*)(lp + 8), 0, 0);
    };
    issue_stage(0, 0);
    __builtin_amdgcn_s_wait_asynccnt(0);
    __syncthreads();
#else
    uint4 r0, r1;
    {
        const uint4* p = (const uint4*)(fbase + (size_t)scl * 512);
        r0 = p[0]; r1 = p[1];
        uint4* q = (uint4*)&lds[0][sset][scl][slane * 16];
        q[0] = r0; q[1] = r1;
    }
    __syncthreads();
#endif

    for (int s = 0; s < NSTAGES; ++s) {
        const int buf = s & 1;
#if USE_ASYNC
        if (s + 1 < NSTAGES) issue_stage(s + 1, buf ^ 1);
#else
        if (s + 1 < NSTAGES) {
            const uint4* p = (const uint4*)(fbase + (size_t)((s + 1) * STAGE + scl) * 512);
            r0 = p[0]; r1 = p[1];
        }
#endif
#pragma unroll
        for (int cl = 0; cl < STAGE; ++cl) {
            const int c = s * STAGE + cl;
            v16h aSv = *(const v16h*)&lds[buf][0][cl][lane * 16];
            v16h aCv = *(const v16h*)&lds[buf][1][cl][lane * 16];
            v16h bv0 = *(const v16h*)(bbase0 + (size_t)STRIDE_ * c);
            v16h bv1 = *(const v16h*)(bbase1 + (size_t)STRIDE_ * c);
            aS0 = __builtin_amdgcn_wmma_f32_16x16x32_f16(false, aSv, false, bv0, (short)0, aS0, false, false);
            aC0 = __builtin_amdgcn_wmma_f32_16x16x32_f16(false, aCv, false, bv0, (short)0, aC0, false, false);
            aS1 = __builtin_amdgcn_wmma_f32_16x16x32_f16(false, aSv, false, bv1, (short)0, aS1, false, false);
            aC1 = __builtin_amdgcn_wmma_f32_16x16x32_f16(false, aCv, false, bv1, (short)0, aC1, false, false);
        }
#if USE_ASYNC
        if (s + 1 < NSTAGES) {
            __builtin_amdgcn_s_wait_asynccnt(0);
            __syncthreads();
        }
#else
        __syncthreads();
        if (s + 1 < NSTAGES) {
            uint4* q = (uint4*)&lds[(s + 1) & 1][sset][scl][slane * 16];
            q[0] = r0; q[1] = r1;
            __syncthreads();
        }
#endif
    }

    // Epilogue: magnitude. C/D layout: VGPR r, lanes 0-15 -> M=r, lanes 16-31 -> M=8+r.
    const int mrow0 = (lane >> 4) << 3;
    const int ncol0 = l0 + (lane & 15);
    const int ncol1 = ncol0 + 16;
#pragma unroll
    for (int r = 0; r < 8; ++r) {
        const int k = m * 16 + mrow0 + r;
        const size_t rowoff = ((size_t)b * KFILT + k) * LOUT;
        if (ncol0 < LOUT) {
            float s = aS0[r], c = aC0[r];
            out[rowoff + ncol0] = sqrtf(s * s + c * c);
        }
        if (ncol1 < LOUT) {
            float s = aS1[r], c = aC1[r];
            out[rowoff + ncol1] = sqrtf(s * s + c * c);
        }
    }
}

// ---------------------------------------------------------------------------
extern "C" void kernel_launch(void* const* d_in, const int* in_sizes, int n_in,
                              void* d_out, int out_size, void* d_ws, size_t ws_size,
                              hipStream_t stream) {
    const float* signal = (const float*)d_in[0];
    const float* wsin   = (const float*)d_in[1];
    const float* wcos   = (const float*)d_in[2];
    float* out = (float*)d_out;

    _Float16* sigH  = (_Float16*)d_ws;                    // 16 * 446464 halves
    _Float16* filtH = sigH + (size_t)NBATCH * TPAD;       // 2 * 15 * 128 * 512 halves

    {
        long total = (long)NBATCH * TPAD;
        int blocks = (int)((total + 255) / 256);
        prep_signal<<<blocks, 256, 0, stream>>>(signal, sigH);
    }
    {
        int total = 2 * MT * NCHUNK * 512;
        prep_filters<<<(total + 255) / 256, 256, 0, stream>>>(wsin, wcos, filtH);
    }
    dim3 grid(NBLK_L, MT, NBATCH);
    spect_wmma<<<grid, 256, 0, stream>>>(sigH, filtH, out);
}
